// ParallelHybridBlock_GF_Ratio_6073083757116
// MI455X (gfx1250) — matrix operand order repaired and checked by reading, not compiled, not run on hardware
//
#include <hip/hip_runtime.h>
#include <cstddef>

// ---------------------------------------------------------------------------
// Types for gfx1250 WMMA (wave32)
// ---------------------------------------------------------------------------
typedef __bf16 bf16_t;
typedef __attribute__((ext_vector_type(16))) __bf16 v16bf;
typedef __attribute__((ext_vector_type(8)))  __bf16 v8bf;
typedef __attribute__((ext_vector_type(8)))  float  v8f;

// ---------------------------------------------------------------------------
// Problem constants (match reference)
// ---------------------------------------------------------------------------
constexpr int Bn   = 2;
constexpr int Tn   = 2048;
constexpr int Hn   = 1024;
constexpr int NHR  = 4;     // RWKV heads
constexpr int HDR  = 256;   // RWKV head dim
constexpr int DIn  = 2048;  // mamba inner
constexpr int NHM  = 32;    // mamba heads
constexpr int HDM  = 64;    // mamba head dim
constexpr int DSn  = 128;   // mamba d_state
constexpr int CKn  = 4;     // conv kernel
constexpr int FFNn = 4096;
constexpr int ZXC  = 2 * DIn + 2 * DSn + NHM; // 4384
constexpr int XBCC = DIn + 2 * DSn;           // 2304
constexpr int ROWS = Bn * Tn;                 // 4096
constexpr float EPSf = 1e-6f;

#define DEVINL __device__ __forceinline__

DEVINL float sigmoidf_(float x) { return 1.f / (1.f + expf(-x)); }
DEVINL float siluf_(float x)    { return x * sigmoidf_(x); }

DEVINL float wave_reduce_sum(float v) {
#pragma unroll
  for (int o = 16; o > 0; o >>= 1) v += __shfl_xor(v, o, 32);
  return v;
}

// blockDim.x <= 1024; `red` must have >= blockDim.x/32 floats
DEVINL float block_reduce_sum(float v, float* red) {
  const int lane = threadIdx.x & 31, wv = threadIdx.x >> 5;
  const int nw = (blockDim.x + 31) >> 5;
  v = wave_reduce_sum(v);
  if (lane == 0) red[wv] = v;
  __syncthreads();
  float t = ((int)threadIdx.x < nw) ? red[threadIdx.x] : 0.f;
  if (wv == 0) t = wave_reduce_sum(t);
  if (threadIdx.x == 0) red[0] = t;
  __syncthreads();
  return red[0];
}

// ---------------------------------------------------------------------------
// Weight prep: W[K][N] f32 (row-major)  ->  Wt[N][K] bf16 (transposed)
// 32x32 tiles through LDS so both global sides stay coalesced.
// ---------------------------------------------------------------------------
__global__ __launch_bounds__(256) void transpose_bf16_kernel(
    const float* __restrict__ W, bf16_t* __restrict__ Wt, int K, int N) {
  __shared__ bf16_t tile[32][33];
  const int tx = threadIdx.x;      // 0..31
  const int ty = threadIdx.y;      // 0..7
  const int k0 = blockIdx.y * 32;
  const int n0 = blockIdx.x * 32;
#pragma unroll
  for (int j = 0; j < 4; ++j) {
    const int r = ty + j * 8;
    const int gk = k0 + r, gn = n0 + tx;
    tile[r][tx] = (gk < K && gn < N) ? (bf16_t)W[(size_t)gk * N + gn] : (bf16_t)0.f;
  }
  __syncthreads();
#pragma unroll
  for (int j = 0; j < 4; ++j) {
    const int r = ty + j * 8;
    const int gn = n0 + r, gk = k0 + tx;
    if (gn < N && gk < K) Wt[(size_t)gn * K + gk] = tile[tx][r];
  }
}

// ---------------------------------------------------------------------------
// bf16 WMMA GEMM: C[M,N] = epi(A[M,K] @ W[K,N]) + (Resid ? Resid : 0)
// A: f32 [M,K] (converted to bf16 while staging).  Wt: bf16 [N,K] (pre-
// transposed).  Requires M%64==0, K%32==0; N may be ragged.
// 64x64 tile/block, BK=32, 8 waves: 4(M) x 2(N), each wave = 16x32 (two WMMA)
// ---------------------------------------------------------------------------
enum { EPI_NONE = 0, EPI_SILU = 1, EPI_DECAY = 2, EPI_GELU = 3 };

template <int EPI>
__global__ __launch_bounds__(256) void gemm_bf16_wmma(
    const float* __restrict__ A, const bf16_t* __restrict__ Wt,
    const float* __restrict__ Resid, float* __restrict__ C,
    int M, int N, int K) {
  __shared__ bf16_t As[64][40];  // [m][k]   (stride 40 halves = 80B, 16B aligned)
  __shared__ bf16_t Bs[64][40];  // [n][k]

  const int tid  = threadIdx.x;
  const int wave = tid >> 5;
  const int lane = tid & 31;
  const int m0 = blockIdx.y * 64;
  const int n0 = blockIdx.x * 64;
  const int wm = wave & 3;        // wave tile row (M)
  const int wn = wave >> 2;       // wave tile col (N)
  const int rIn = lane & 15;      // lane row/col within 16
  const int hl  = lane >> 4;      // lane half (K phase select)

  v8f c0 = {0.f, 0.f, 0.f, 0.f, 0.f, 0.f, 0.f, 0.f};
  v8f c1 = c0;

  const int lr = tid >> 2;        // 0..63 : row (A) / col (B) being staged
  const int lk = (tid & 3) * 8;   // 0,8,16,24 : k sub-offset
  const int gm = m0 + lr;         // always < M (M%64==0)
  const int gn = n0 + lr;         // may be >= N (ragged)
  const bool bok = (gn < N);
  const float*  aRow = A  + (size_t)gm * K;
  const bf16_t* bRow = Wt + (size_t)(bok ? gn : 0) * K;

  // ---- prologue: stage tile k0=0 into registers ----
  v8bf areg, breg;
  {
    const float4 a0 = *(const float4*)(aRow + lk);
    const float4 a1 = *(const float4*)(aRow + lk + 4);
    areg[0] = (bf16_t)a0.x; areg[1] = (bf16_t)a0.y;
    areg[2] = (bf16_t)a0.z; areg[3] = (bf16_t)a0.w;
    areg[4] = (bf16_t)a1.x; areg[5] = (bf16_t)a1.y;
    areg[6] = (bf16_t)a1.z; areg[7] = (bf16_t)a1.w;
    breg = bok ? *(const v8bf*)(bRow + lk) : (v8bf)(bf16_t)0.f;
  }

  for (int k0 = 0; k0 < K; k0 += 32) {
    // ---- commit staged registers to LDS (single b128 each) ----
    *(v8bf*)&As[lr][lk] = areg;
    *(v8bf*)&Bs[lr][lk] = breg;
    __syncthreads();

    // ---- issue next tile's global loads (overlaps ds/wmma below) ----
    const int kn = k0 + 32;
    if (kn < K) {
      const float4 a0 = *(const float4*)(aRow + kn + lk);
      const float4 a1 = *(const float4*)(aRow + kn + lk + 4);
      areg[0] = (bf16_t)a0.x; areg[1] = (bf16_t)a0.y;
      areg[2] = (bf16_t)a0.z; areg[3] = (bf16_t)a0.w;
      areg[4] = (bf16_t)a1.x; areg[5] = (bf16_t)a1.y;
      areg[6] = (bf16_t)a1.z; areg[7] = (bf16_t)a1.w;
      breg = bok ? *(const v8bf*)(bRow + kn + lk) : (v8bf)(bf16_t)0.f;
      if (kn + 32 < K) {  // global_prefetch_b8 for the tile after next
        __builtin_prefetch(aRow + kn + 32 + lk, 0, 1);
        __builtin_prefetch(bRow + kn + 32 + lk, 0, 1);
      }
    }

    // ---- gather WMMA fragments (contiguous k, same order for A and B) ----
    // A 16x32 bf16: lane L -> row M=L%16; half j: K = hl*16 + j (16 contiguous)
    union { v16bf v; v8bf h[2]; } af, b0f, b1f;
    const bf16_t* pa = &As[wm * 16 + rIn][0];
    af.h[0] = *(const v8bf*)(pa + hl * 16);
    af.h[1] = *(const v8bf*)(pa + hl * 16 + 8);
    // B 32x16 bf16: lane L -> col N=L%16; half j: K = hl*16 + j (16 contiguous)
    const bf16_t* pb0 = &Bs[wn * 32 + rIn][0];
    b0f.h[0] = *(const v8bf*)(pb0 + hl * 16);
    b0f.h[1] = *(const v8bf*)(pb0 + hl * 16 + 8);
    const bf16_t* pb1 = &Bs[wn * 32 + 16 + rIn][0];
    b1f.h[0] = *(const v8bf*)(pb1 + hl * 16);
    b1f.h[1] = *(const v8bf*)(pb1 + hl * 16 + 8);

    c0 = __builtin_amdgcn_wmma_f32_16x16x32_bf16(false, af.v, false, b0f.v,
                                                 (short)0, c0, false, false);
    c1 = __builtin_amdgcn_wmma_f32_16x16x32_bf16(false, af.v, false, b1f.v,
                                                 (short)0, c1, false, false);
    __syncthreads();
  }

  // ---- epilogue + store. C/D layout: lane -> col N=L%16, vgpr v -> row v+8*hl
  const int gn0 = n0 + wn * 32 + rIn;
  const int gn1 = gn0 + 16;
#pragma unroll
  for (int v = 0; v < 8; ++v) {
    const int gmo = m0 + wm * 16 + v + hl * 8;
    float x0 = c0[v], x1 = c1[v];
    if (EPI == EPI_SILU) {
      x0 = siluf_(x0); x1 = siluf_(x1);
    } else if (EPI == EPI_DECAY) {
      x0 = expf(-expf(x0)); x1 = expf(-expf(x1));
    } else if (EPI == EPI_GELU) {
      x0 = 0.5f * x0 * (1.f + erff(x0 * 0.70710678118654752f));
      x1 = 0.5f * x1 * (1.f + erff(x1 * 0.70710678118654752f));
    }
    const size_t i0 = (size_t)gmo * N + gn0;
    const size_t i1 = (size_t)gmo * N + gn1;
    if (gn0 < N) C[i0] = x0 + (Resid ? Resid[i0] : 0.f);
    if (gn1 < N) C[i1] = x1 + (Resid ? Resid[i1] : 0.f);
  }
}

// ---------------------------------------------------------------------------
// RMSNorm: y[row] = w * x[row] * rsqrt(mean(x^2) + eps)   (one block per row)
// ---------------------------------------------------------------------------
__global__ __launch_bounds__(256) void rmsnorm_kernel(
    const float* __restrict__ x, const float* __restrict__ w,
    float* __restrict__ y, int D) {
  __shared__ float red[32];
  const size_t row = blockIdx.x;
  const float* xr = x + row * D;
  float ss = 0.f;
  for (int i = threadIdx.x; i < D; i += blockDim.x) { float v = xr[i]; ss += v * v; }
  const float tot = block_reduce_sum(ss, red);
  const float inv = rsqrtf(tot / (float)D + EPSf);
  for (int i = threadIdx.x; i < D; i += blockDim.x) y[row * D + i] = w[i] * xr[i] * inv;
}

// ---------------------------------------------------------------------------
// Causal depthwise conv (CK=4) + bias + SiLU over xBC channels of zxbcdt
// ---------------------------------------------------------------------------
__global__ __launch_bounds__(256) void conv_silu_kernel(
    const float* __restrict__ zx, const float* __restrict__ cw,
    const float* __restrict__ cb, float* __restrict__ xbc) {
  const int idx = blockIdx.x * 256 + threadIdx.x;
  if (idx >= ROWS * XBCC) return;
  const int c = idx % XBCC;
  const int bt = idx / XBCC;
  const int t = bt % Tn, b = bt / Tn;
  float acc = cb[c];
#pragma unroll
  for (int i = 0; i < CKn; ++i) {
    const int tt = t - (CKn - 1) + i;
    if (tt >= 0) acc += zx[((size_t)(b * Tn + tt)) * ZXC + DIn + c] * cw[c * CKn + i];
  }
  xbc[idx] = siluf_(acc);
}

// dt = softplus(zxbcdt[..., 4352:4384] + dt_bias)
__global__ __launch_bounds__(256) void dt_softplus_kernel(
    const float* __restrict__ zx, const float* __restrict__ dtb,
    float* __restrict__ dt) {
  const int idx = blockIdx.x * 256 + threadIdx.x;
  if (idx >= ROWS * NHM) return;
  const int h = idx % NHM;
  const size_t bt = idx / NHM;
  const float x = zx[bt * ZXC + (2 * DIn + 2 * DSn) + h] + dtb[h];
  dt[idx] = (x > 20.f) ? x : log1pf(expf(x));
}

// ---------------------------------------------------------------------------
// RWKV6 recurrence. One block per (b, head); 1024 threads = 32x32 grid of
// (kgroup, vgroup); each thread holds an 8x8 register tile of the 256x256
// state.  o_t = r . (S + u*k v^T), then S = w*S + k v^T.
// ---------------------------------------------------------------------------
__global__ __launch_bounds__(1024) void rwkv_scan_kernel(
    const float* __restrict__ r, const float* __restrict__ k,
    const float* __restrict__ v, const float* __restrict__ wdec,
    const float* __restrict__ u, float* __restrict__ oraw) {
  __shared__ float shr[4][HDR];      // r,k,v,w for current t
  __shared__ float red[32][HDR];     // k-group partial sums
  const int tid = threadIdx.x;
  const int kg = tid >> 5;           // 0..31
  const int vg = tid & 31;           // 0..31
  const int head = blockIdx.x % NHR;
  const int b = blockIdx.x / NHR;

  float S[8][8];
#pragma unroll
  for (int i = 0; i < 8; ++i)
#pragma unroll
    for (int j = 0; j < 8; ++j) S[i][j] = 0.f;

  float uloc[8];
#pragma unroll
  for (int i = 0; i < 8; ++i) uloc[i] = u[head * HDR + kg * 8 + i];

  for (int t = 0; t < Tn; ++t) {
    const size_t base = ((size_t)(b * Tn + t)) * Hn + head * HDR;
    __syncthreads();
    {
      const int arr = tid >> 8;   // 0..3
      const int i = tid & 255;
      float val;
      if      (arr == 0) val = r[base + i];
      else if (arr == 1) val = k[base + i];
      else if (arr == 2) val = v[base + i];
      else               val = wdec[base + i];
      shr[arr][i] = val;
    }
    __syncthreads();

    float opart[8];
#pragma unroll
    for (int vv = 0; vv < 8; ++vv) opart[vv] = 0.f;
#pragma unroll
    for (int kk = 0; kk < 8; ++kk) {
      const int kidx = kg * 8 + kk;
      const float rk = shr[0][kidx];
      const float kv_k = shr[1][kidx];
      const float wk = shr[3][kidx];
      const float uk = uloc[kk];
#pragma unroll
      for (int vv = 0; vv < 8; ++vv) {
        const float vval = shr[2][vg * 8 + vv];
        const float kv = kv_k * vval;
        const float s = S[kk][vv];
        opart[vv] += rk * (s + uk * kv);
        S[kk][vv] = wk * s + kv;
      }
    }
#pragma unroll
    for (int vv = 0; vv < 8; ++vv) red[kg][vg * 8 + vv] = opart[vv];
    __syncthreads();
    if (tid < HDR) {
      float acc = 0.f;
#pragma unroll
      for (int g = 0; g < 32; ++g) acc += red[g][tid];
      oraw[base + tid] = acc;
    }
  }
}

// ---------------------------------------------------------------------------
// Mamba2 SSD recurrence.  One block per (b, head); thread (hd, q) owns 32
// d_state entries; y reduced over 4 lanes via shuffle; +D*x fused.
// ---------------------------------------------------------------------------
__global__ __launch_bounds__(256) void mamba_scan_kernel(
    const float* __restrict__ xbc, const float* __restrict__ dt,
    const float* __restrict__ A_log, const float* __restrict__ D_m,
    float* __restrict__ y) {
  __shared__ float sB[DSn], sC[DSn], sx[HDM];
  const int tid = threadIdx.x;
  const int hd = tid >> 2, q = tid & 3;
  const int head = blockIdx.x % NHM;
  const int b = blockIdx.x / NHM;
  const float Av = -expf(A_log[head]);
  const float Dv = D_m[head];

  float h[32];
#pragma unroll
  for (int j = 0; j < 32; ++j) h[j] = 0.f;

  for (int t = 0; t < Tn; ++t) {
    const size_t row = (size_t)(b * Tn + t);
    const size_t ro = row * XBCC;
    __syncthreads();
    if (tid < DSn) {
      sB[tid] = xbc[ro + DIn + tid];
      sC[tid] = xbc[ro + DIn + DSn + tid];
    }
    if (tid < HDM) sx[tid] = xbc[ro + (size_t)head * HDM + tid];
    __syncthreads();

    const float dtv = dt[row * NHM + head];
    const float decay = expf(dtv * Av);
    const float coef = dtv * sx[hd];
    float yp = 0.f;
#pragma unroll
    for (int j = 0; j < 32; ++j) {
      const int d = q * 32 + j;
      const float hv = h[j] * decay + coef * sB[d];
      h[j] = hv;
      yp += hv * sC[d];
    }
    yp += __shfl_xor(yp, 1, 32);
    yp += __shfl_xor(yp, 2, 32);
    if (q == 0) y[row * DIn + head * HDM + hd] = yp + Dv * sx[hd];
  }
}

// RWKV per-head RMS (group norm) * gn_w * silu-gate
__global__ __launch_bounds__(256) void rwkv_gn_gate_kernel(
    const float* __restrict__ oraw, const float* __restrict__ gnw,
    const float* __restrict__ g, float* __restrict__ out) {
  __shared__ float red[32];
  const int head = blockIdx.x % NHR;
  const size_t bt = blockIdx.x / NHR;
  const size_t idx = bt * Hn + (size_t)head * HDR + threadIdx.x;
  const float o = oraw[idx];
  const float tot = block_reduce_sum(o * o, red);
  const float inv = rsqrtf(tot / (float)HDR + EPSf);
  out[idx] = o * inv * gnw[head * HDR + threadIdx.x] * g[idx];
}

// Mamba gated RMSNorm: out = m_norm_w * (y*silu(z)) * rsqrt(mean+eps)  (in place OK)
__global__ __launch_bounds__(256) void mamba_rms_kernel(
    const float* __restrict__ y, const float* __restrict__ zx,
    const float* __restrict__ nw, float* __restrict__ out) {
  __shared__ float red[32];
  const size_t row = blockIdx.x;
  float vals[8];
  float ss = 0.f;
#pragma unroll
  for (int j = 0; j < 8; ++j) {
    const int i = threadIdx.x + j * 256;
    const float z = zx[row * ZXC + i];
    const float v = y[row * DIn + i] * siluf_(z);
    vals[j] = v;
    ss += v * v;
  }
  const float tot = block_reduce_sum(ss, red);
  const float inv = rsqrtf(tot / (float)DIn + EPSf);
#pragma unroll
  for (int j = 0; j < 8; ++j) {
    const int i = threadIdx.x + j * 256;
    out[row * DIn + i] = nw[i] * vals[j] * inv;
  }
}

// gate = sigmoid([o_r|o_m] @ gate_w + gate_b); x1 = x + g*o_r + (1-g)*o_m
__global__ __launch_bounds__(256) void gate_combine_kernel(
    const float* __restrict__ x, const float* __restrict__ o_r,
    const float* __restrict__ o_m, const float* __restrict__ gw,
    const float* __restrict__ gbias, float* __restrict__ x1) {
  __shared__ float red[32];
  const size_t row = blockIdx.x;
  float dot = 0.f;
#pragma unroll
  for (int j = 0; j < 4; ++j) {
    const int i = threadIdx.x + j * 256;
    dot += o_r[row * Hn + i] * gw[i] + o_m[row * Hn + i] * gw[Hn + i];
  }
  const float tot = block_reduce_sum(dot, red);
  const float gate = sigmoidf_(tot + gbias[0]);
#pragma unroll
  for (int j = 0; j < 4; ++j) {
    const int i = threadIdx.x + j * 256;
    x1[row * Hn + i] =
        x[row * Hn + i] + gate * o_r[row * Hn + i] + (1.f - gate) * o_m[row * Hn + i];
  }
}

// ---------------------------------------------------------------------------
// Launcher
// ---------------------------------------------------------------------------
extern "C" void kernel_launch(void* const* d_in, const int* in_sizes, int n_in,
                              void* d_out, int out_size, void* d_ws, size_t ws_size,
                              hipStream_t stream) {
  (void)in_sizes; (void)n_in; (void)out_size; (void)ws_size;
  const float* x        = (const float*)d_in[0];
  const float* ln_w     = (const float*)d_in[1];
  const float* r_w      = (const float*)d_in[2];
  const float* k_w      = (const float*)d_in[3];
  const float* v_w      = (const float*)d_in[4];
  const float* g_w      = (const float*)d_in[5];
  const float* dw_w     = (const float*)d_in[6];
  const float* u        = (const float*)d_in[7];
  const float* gn_w     = (const float*)d_in[8];
  const float* o_w      = (const float*)d_in[9];
  const float* in_proj  = (const float*)d_in[10];
  const float* conv_w   = (const float*)d_in[11];
  const float* conv_b   = (const float*)d_in[12];
  const float* dt_bias  = (const float*)d_in[13];
  const float* A_log    = (const float*)d_in[14];
  const float* D_m      = (const float*)d_in[15];
  const float* m_norm_w = (const float*)d_in[16];
  const float* out_proj = (const float*)d_in[17];
  const float* gate_w   = (const float*)d_in[18];
  const float* gate_b   = (const float*)d_in[19];
  const float* ffn_ln_w = (const float*)d_in[20];
  const float* ffn_w1   = (const float*)d_in[21];
  const float* ffn_w2   = (const float*)d_in[22];
  float* outp = (float*)d_out;

  float* ws = (float*)d_ws;
  size_t off = 0;
  auto alloc = [&](size_t n) { float* p = ws + off; off += n; return p; };
  float* nx   = alloc((size_t)ROWS * Hn);
  float* rb   = alloc((size_t)ROWS * Hn);   // rb..gbuf contiguous: reused as ffnh
  float* kb   = alloc((size_t)ROWS * Hn);
  float* vb   = alloc((size_t)ROWS * Hn);
  float* gbuf = alloc((size_t)ROWS * Hn);
  float* wdec = alloc((size_t)ROWS * Hn);   // reused as o_gated after scan
  float* oraw = alloc((size_t)ROWS * Hn);
  float* o_r  = alloc((size_t)ROWS * Hn);
  float* zx   = alloc((size_t)ROWS * ZXC);
  float* xbc  = alloc((size_t)ROWS * XBCC);
  float* dtb  = alloc((size_t)ROWS * NHM);
  float* yb   = alloc((size_t)ROWS * DIn);
  float* o_m  = alloc((size_t)ROWS * Hn);
  float* x1   = alloc((size_t)ROWS * Hn);
  // bf16 transposed-weight scratch (reused per GEMM; max = in_proj 4384*1024)
  bf16_t* wt  = (bf16_t*)(ws + off);
  float* ffnh  = rb;    // [ROWS, 4096] aliases rb..gbuf (free by FFN stage)
  float* ogate = wdec;  // aliases wdec (free after RWKV scan)

  const dim3 blk(256);
  const dim3 tblk(32, 8);
  auto ggrid = [](int M, int N) { return dim3((N + 63) / 64, (M + 63) / 64); };
  auto tgrid = [](int K, int N) { return dim3((N + 31) / 32, (K + 31) / 32); };

  // 1) pre-norm
  rmsnorm_kernel<<<ROWS, blk, 0, stream>>>(x, ln_w, nx, Hn);

  // 2) RWKV projections (WMMA) with fused activations; weight prep before each
  transpose_bf16_kernel<<<tgrid(Hn, Hn), tblk, 0, stream>>>(r_w, wt, Hn, Hn);
  gemm_bf16_wmma<EPI_NONE ><<<ggrid(ROWS, Hn), blk, 0, stream>>>(nx, wt, nullptr, rb,   ROWS, Hn, Hn);
  transpose_bf16_kernel<<<tgrid(Hn, Hn), tblk, 0, stream>>>(k_w, wt, Hn, Hn);
  gemm_bf16_wmma<EPI_NONE ><<<ggrid(ROWS, Hn), blk, 0, stream>>>(nx, wt, nullptr, kb,   ROWS, Hn, Hn);
  transpose_bf16_kernel<<<tgrid(Hn, Hn), tblk, 0, stream>>>(v_w, wt, Hn, Hn);
  gemm_bf16_wmma<EPI_NONE ><<<ggrid(ROWS, Hn), blk, 0, stream>>>(nx, wt, nullptr, vb,   ROWS, Hn, Hn);
  transpose_bf16_kernel<<<tgrid(Hn, Hn), tblk, 0, stream>>>(g_w, wt, Hn, Hn);
  gemm_bf16_wmma<EPI_SILU ><<<ggrid(ROWS, Hn), blk, 0, stream>>>(nx, wt, nullptr, gbuf, ROWS, Hn, Hn);
  transpose_bf16_kernel<<<tgrid(Hn, Hn), tblk, 0, stream>>>(dw_w, wt, Hn, Hn);
  gemm_bf16_wmma<EPI_DECAY><<<ggrid(ROWS, Hn), blk, 0, stream>>>(nx, wt, nullptr, wdec, ROWS, Hn, Hn);

  // 3) Mamba input projection (WMMA)
  transpose_bf16_kernel<<<tgrid(Hn, ZXC), tblk, 0, stream>>>(in_proj, wt, Hn, ZXC);
  gemm_bf16_wmma<EPI_NONE><<<ggrid(ROWS, ZXC), blk, 0, stream>>>(nx, wt, nullptr, zx, ROWS, ZXC, Hn);
  conv_silu_kernel<<<(ROWS * XBCC + 255) / 256, blk, 0, stream>>>(zx, conv_w, conv_b, xbc);
  dt_softplus_kernel<<<(ROWS * NHM + 255) / 256, blk, 0, stream>>>(zx, dt_bias, dtb);

  // 4) recurrences
  rwkv_scan_kernel<<<Bn * NHR, 1024, 0, stream>>>(rb, kb, vb, wdec, u, oraw);
  mamba_scan_kernel<<<Bn * NHM, blk, 0, stream>>>(xbc, dtb, A_log, D_m, yb);

  // 5) RWKV output path: group-norm * gate, then o_w (WMMA)
  rwkv_gn_gate_kernel<<<ROWS * NHR, blk, 0, stream>>>(oraw, gn_w, gbuf, ogate);
  transpose_bf16_kernel<<<tgrid(Hn, Hn), tblk, 0, stream>>>(o_w, wt, Hn, Hn);
  gemm_bf16_wmma<EPI_NONE><<<ggrid(ROWS, Hn), blk, 0, stream>>>(ogate, wt, nullptr, o_r, ROWS, Hn, Hn);

  // 6) Mamba output path: gated RMSNorm (in place), then out_proj (WMMA)
  mamba_rms_kernel<<<ROWS, blk, 0, stream>>>(yb, zx, m_norm_w, yb);
  transpose_bf16_kernel<<<tgrid(DIn, Hn), tblk, 0, stream>>>(out_proj, wt, DIn, Hn);
  gemm_bf16_wmma<EPI_NONE><<<ggrid(ROWS, Hn), blk, 0, stream>>>(yb, wt, nullptr, o_m, ROWS, Hn, DIn);

  // 7) sigmoid gate mix + residual
  gate_combine_kernel<<<ROWS, blk, 0, stream>>>(x, o_r, o_m, gate_w, gate_b, x1);

  // 8) FFN: rms -> w1 (+exact GELU, WMMA) -> w2 (+residual, WMMA) -> d_out
  rmsnorm_kernel<<<ROWS, blk, 0, stream>>>(x1, ffn_ln_w, nx, Hn);
  transpose_bf16_kernel<<<tgrid(Hn, FFNn), tblk, 0, stream>>>(ffn_w1, wt, Hn, FFNn);
  gemm_bf16_wmma<EPI_GELU><<<ggrid(ROWS, FFNn), blk, 0, stream>>>(nx, wt, nullptr, ffnh, ROWS, FFNn, Hn);
  transpose_bf16_kernel<<<tgrid(FFNn, Hn), tblk, 0, stream>>>(ffn_w2, wt, FFNn, Hn);
  gemm_bf16_wmma<EPI_NONE><<<ggrid(ROWS, Hn), blk, 0, stream>>>(ffnh, wt, x1, outp, ROWS, Hn, FFNn);
}
